// PointNetSetAbstractionMsg_42631845380156
// MI455X (gfx1250) — matrix operand order repaired and checked by reading, not compiled, not run on hardware
//
#include <hip/hip_runtime.h>

// ---------------------------------------------------------------------------
// PointNet++ MSG set-abstraction for MI455X (gfx1250, wave32, WMMA).
// Stages: FPS -> gather centers -> dual-radius ball query (ballot scan) ->
//         folded-BN MLP via V_WMMA_F32_16X16X32_F16 + max-pool.
// ---------------------------------------------------------------------------

typedef _Float16 v16h __attribute__((ext_vector_type(16)));
typedef _Float16 v8h  __attribute__((ext_vector_type(8)));
typedef float    v8f  __attribute__((ext_vector_type(8)));

#define NBATCH 8
#define NPTS   16384
#define NPOINT 1024

// ---------------------------------------------------------------------------
// 1) Farthest point sampling: one 1024-thread block per batch.
//    dist kept in registers (16 pts/thread), xyz re-read from L2-hot global.
// ---------------------------------------------------------------------------
__global__ void fps_kernel(const float* __restrict__ xyz, int* __restrict__ fps_idx) {
    const int b   = blockIdx.x;
    const int tid = threadIdx.x;
    const int lane = tid & 31;
    const int wid  = tid >> 5;
    const float* xp = xyz + (size_t)b * 3 * NPTS;

    __shared__ float s_bd[32];
    __shared__ int   s_bi[32];
    __shared__ float s_c[3];
    __shared__ int   s_far;

    float dist[16];
#pragma unroll
    for (int i = 0; i < 16; ++i) dist[i] = 1.0e10f;

    if (tid == 0) {
        s_far = 0;
        s_c[0] = xp[0]; s_c[1] = xp[NPTS]; s_c[2] = xp[2 * NPTS];
    }
    __syncthreads();

    for (int it = 0; it < NPOINT; ++it) {
        if (tid == 0) fps_idx[b * NPOINT + it] = s_far;   // record BEFORE update
        const float cx = s_c[0], cy = s_c[1], cz = s_c[2];

        float bd = -1.0f; int bi = 0;
#pragma unroll
        for (int i = 0; i < 16; ++i) {
            const int n = tid + i * 1024;
            const float dx = xp[n] - cx;
            const float dy = xp[NPTS + n] - cy;
            const float dz = xp[2 * NPTS + n] - cz;
            const float d  = dx * dx + dy * dy + dz * dz;
            const float nd = fminf(dist[i], d);
            dist[i] = nd;
            if (nd > bd || (nd == bd && n < bi)) { bd = nd; bi = n; }
        }
        // wave32 reduction (argmax, tie -> smaller index = first occurrence)
#pragma unroll
        for (int off = 16; off > 0; off >>= 1) {
            const float od = __shfl_xor(bd, off);
            const int   oi = __shfl_xor(bi, off);
            if (od > bd || (od == bd && oi < bi)) { bd = od; bi = oi; }
        }
        if (lane == 0) { s_bd[wid] = bd; s_bi[wid] = bi; }
        __syncthreads();
        if (tid == 0) {
            float gb = s_bd[0]; int gi = s_bi[0];
            for (int w = 1; w < 32; ++w) {
                const float od = s_bd[w]; const int oi = s_bi[w];
                if (od > gb || (od == gb && oi < gi)) { gb = od; gi = oi; }
            }
            s_far = gi;
            s_c[0] = xp[gi]; s_c[1] = xp[NPTS + gi]; s_c[2] = xp[2 * NPTS + gi];
        }
        __syncthreads();
    }
}

// ---------------------------------------------------------------------------
// 2) Gather centers -> d_out (B,3,S) and ws newxyz (B,S,3)
// ---------------------------------------------------------------------------
__global__ void gather_centers_kernel(const float* __restrict__ xyz,
                                      const int* __restrict__ fps_idx,
                                      float* __restrict__ out_xyz,  // (B,3,S)
                                      float* __restrict__ nx) {     // (B*S,3)
    const int t = blockIdx.x * blockDim.x + threadIdx.x;
    if (t >= NBATCH * NPOINT) return;
    const int b = t >> 10, s = t & 1023;
    const int idx = fps_idx[t];
    const float* xp = xyz + (size_t)b * 3 * NPTS;
#pragma unroll
    for (int c = 0; c < 3; ++c) {
        const float v = xp[c * NPTS + idx];
        out_xyz[(size_t)b * 3 * NPOINT + c * NPOINT + s] = v;
        nx[(size_t)t * 3 + c] = v;
    }
}

// ---------------------------------------------------------------------------
// 3) Dual-radius ball query: one wave per (b,s) group.
//    Ballot over 32-point chunks preserves ascending-index first-K semantics.
// ---------------------------------------------------------------------------
__global__ void query_ball_kernel(const float* __restrict__ xyz,
                                  const float* __restrict__ nx,
                                  int* __restrict__ g0,   // (B*S,16)
                                  int* __restrict__ g1) { // (B*S,32)
    const int lane = threadIdx.x & 31;
    const int gs = blockIdx.x * 8 + (threadIdx.x >> 5);   // group id 0..8191
    const int b = gs >> 10;
    const float* xp = xyz + (size_t)b * 3 * NPTS;
    const float cx = nx[(size_t)gs * 3 + 0];
    const float cy = nx[(size_t)gs * 3 + 1];
    const float cz = nx[(size_t)gs * 3 + 2];
    const float r0sq = 0.05f * 0.05f;
    const float r1sq = 0.10f * 0.10f;
    int* __restrict__ o0 = g0 + (size_t)gs * 16;
    int* __restrict__ o1 = g1 + (size_t)gs * 32;
    const unsigned lmask = (1u << lane) - 1u;

    int cnt0 = 0, cnt1 = 0, first0 = -1, first1 = -1;
    for (int n0 = 0; n0 < NPTS; n0 += 32) {
        const int n = n0 + lane;
        const float dx = xp[n] - cx;
        const float dy = xp[NPTS + n] - cy;
        const float dz = xp[2 * NPTS + n] - cz;
        const float d = dx * dx + dy * dy + dz * dz;
        const unsigned m0 = (unsigned)__ballot(d <= r0sq);
        const unsigned m1 = (unsigned)__ballot(d <= r1sq);
        if (m0) {
            if (first0 < 0) first0 = n0 + __builtin_ctz(m0);
            const int pos = cnt0 + __popc(m0 & lmask);
            if ((d <= r0sq) && pos < 16) o0[pos] = n;
            cnt0 += __popc(m0);
        }
        if (m1) {
            if (first1 < 0) first1 = n0 + __builtin_ctz(m1);
            const int pos = cnt1 + __popc(m1 & lmask);
            if ((d <= r1sq) && pos < 32) o1[pos] = n;
            cnt1 += __popc(m1);
        }
        if (cnt0 >= 16 && cnt1 >= 32) break;   // wave-uniform
    }
    if (first0 < 0) first0 = 0;
    if (first1 < 0) first1 = 0;
    if (lane < 16 && lane >= cnt0) o0[lane] = first0;
    if (lane < 32 && lane >= cnt1) o1[lane] = first1;
}

// ---------------------------------------------------------------------------
// 4) Fold BN into weights: W' = W * gamma/sqrt(var+eps), b' = (b-mean)*s+beta
//    Stored padded 32x32 f16 per layer in B-matrix-friendly [out][in] rows.
// ---------------------------------------------------------------------------
struct Params36 { const float* p[36]; };

__global__ void fold_weights_kernel(Params36 P, _Float16* __restrict__ Wf,
                                    float* __restrict__ bf) {
    const int t = threadIdx.x;          // 192 threads: 6 layers x 32 outs
    const int l = t >> 5, n = t & 31;
    const int li = l % 3;
    const int INCH[3]  = {9, 6, 32};
    const int OUTCH[3] = {6, 32, 32};
    const int in  = INCH[li];
    const int out = OUTCH[li];
    const float* W     = P.p[l * 6 + 0];
    const float* bias  = P.p[l * 6 + 1];
    const float* gamma = P.p[l * 6 + 2];
    const float* beta  = P.p[l * 6 + 3];
    const float* mean  = P.p[l * 6 + 4];
    const float* var   = P.p[l * 6 + 5];
    float scale = 0.0f, bb = 0.0f;
    if (n < out) {
        scale = gamma[n] * rsqrtf(var[n] + 1e-5f);
        bb = (bias[n] - mean[n]) * scale + beta[n];
    }
    bf[l * 32 + n] = bb;
    for (int k = 0; k < 32; ++k)
        Wf[(size_t)l * 1024 + n * 32 + k] =
            (_Float16)((n < out && k < in) ? W[n * in + k] * scale : 0.0f);
}

// ---------------------------------------------------------------------------
// 5) MLP + maxpool via WMMA. One wave per group; 8 waves per block.
//    LDS activation tiles [2 buffers][32 rows][32 ch] f16, A-layout DS loads.
// ---------------------------------------------------------------------------
__global__ void mlp_wmma_kernel(const int* __restrict__ gidx,
                                const float* __restrict__ xyz,
                                const float* __restrict__ pts,
                                const float* __restrict__ nx,
                                const _Float16* __restrict__ Wf,
                                const float* __restrict__ bf,
                                float* __restrict__ outp,   // (B,64,S)
                                int branch, int K, int mtiles) {
    __shared__ __align__(64) _Float16 Xs[8][2][32][32];
    const int lane = threadIdx.x & 31;
    const int wid  = threadIdx.x >> 5;
    const int gs   = blockIdx.x * 8 + wid;      // 0..8191
    const int b    = gs >> 10, s = gs & 1023;

    // zero both activation buffers (each lane owns one row of each)
    {
        int* r0 = (int*)&Xs[wid][0][lane][0];
        int* r1 = (int*)&Xs[wid][1][lane][0];
#pragma unroll
        for (int j = 0; j < 16; ++j) { r0[j] = 0; r1[j] = 0; }
    }
    // gather features: lane k -> sample k: [points(6) | rel_xyz(3)]
    if (lane < K) {
        const int idx = gidx[(size_t)gs * K + lane];
        const float* xp = xyz + (size_t)b * 3 * NPTS;
        const float* pp = pts + (size_t)b * 6 * NPTS;
        const float cx = nx[(size_t)gs * 3 + 0];
        const float cy = nx[(size_t)gs * 3 + 1];
        const float cz = nx[(size_t)gs * 3 + 2];
#pragma unroll
        for (int c = 0; c < 6; ++c)
            Xs[wid][0][lane][c] = (_Float16)pp[c * NPTS + idx];
        Xs[wid][0][lane][6] = (_Float16)(xp[idx] - cx);
        Xs[wid][0][lane][7] = (_Float16)(xp[NPTS + idx] - cy);
        Xs[wid][0][lane][8] = (_Float16)(xp[2 * NPTS + idx] - cz);
    }
    __syncthreads();

    const int klo   = (lane < 16) ? 0 : 16;   // A-layout K offset per half-wave
    const int kbase = (lane < 16) ? 0 : 16;   // B-layout K offset per half-wave
    const int col   = lane & 15;
    float vmax[2] = {0.0f, 0.0f};             // relu output >= 0

    for (int l = 0; l < 3; ++l) {
        const int nt  = (l == 0) ? 1 : 2;
        const int src = l & 1;                // 0,1,0
        const int dst = src ^ 1;
        const int lidx = branch * 3 + l;
        for (int ntile = 0; ntile < nt; ++ntile) {
            const float bb = bf[lidx * 32 + ntile * 16 + col];
            // B operand: lane holds column (out ch), 16 contiguous K halves
            const v16h bmat = *(const v16h*)(Wf + ((size_t)lidx * 32 +
                                                   ntile * 16 + col) * 32 + kbase);
            for (int mtile = 0; mtile < mtiles; ++mtile) {
                const int row = mtile * 16 + col;
                const v8h lo = *(const v8h*)&Xs[wid][src][row][klo];
                const v8h hi = *(const v8h*)&Xs[wid][src][row][klo + 8];
                v16h a;
#pragma unroll
                for (int i = 0; i < 8; ++i) { a[i] = lo[i]; a[i + 8] = hi[i]; }
                v8f c;
#pragma unroll
                for (int i = 0; i < 8; ++i) c[i] = bb;
                c = __builtin_amdgcn_wmma_f32_16x16x32_f16(
                        false, a, false, bmat, (short)0, c, false, false);
                if (l < 2) {
                    const int rowbase = mtile * 16 + ((lane < 16) ? 0 : 8);
                    const int cs = ntile * 16 + col;
#pragma unroll
                    for (int i = 0; i < 8; ++i)
                        Xs[wid][dst][rowbase + i][cs] =
                            (_Float16)fmaxf(c[i], 0.0f);
                } else {
#pragma unroll
                    for (int i = 0; i < 8; ++i)
                        vmax[ntile] = fmaxf(vmax[ntile], fmaxf(c[i], 0.0f));
                }
            }
        }
        __syncthreads();
    }
    // combine the two half-wave row groups, then write (B,64,S)
#pragma unroll
    for (int ntile = 0; ntile < 2; ++ntile) {
        float m = vmax[ntile];
        m = fmaxf(m, __shfl_xor(m, 16));
        if (lane < 16) {
            const int ch = branch * 32 + ntile * 16 + lane;
            outp[((size_t)b * 64 + ch) * NPOINT + s] = m;
        }
    }
}

// ---------------------------------------------------------------------------
extern "C" void kernel_launch(void* const* d_in, const int* in_sizes, int n_in,
                              void* d_out, int out_size, void* d_ws, size_t ws_size,
                              hipStream_t stream) {
    const float* xyz = (const float*)d_in[0];   // (8,3,16384)
    const float* pts = (const float*)d_in[1];   // (8,6,16384)

    // workspace carve (all 64B aligned)
    char* ws = (char*)d_ws;
    int*       fps = (int*)(ws + 0);              //  32 KB
    float*     nx  = (float*)(ws + 32768);        //  96 KB (B*S,3)
    int*       g0  = (int*)(ws + 131072);         // 512 KB (B*S,16)
    int*       g1  = (int*)(ws + 655360);         //   1 MB (B*S,32)
    _Float16*  Wf  = (_Float16*)(ws + 1703936);   //  12 KB
    float*     bf  = (float*)(ws + 1716224);      // 768 B

    float* out_xyz = (float*)d_out;                         // (8,3,1024)
    float* out_pts = (float*)d_out + NBATCH * 3 * NPOINT;   // (8,64,1024)

    fps_kernel<<<NBATCH, 1024, 0, stream>>>(xyz, fps);
    gather_centers_kernel<<<(NBATCH * NPOINT + 255) / 256, 256, 0, stream>>>(
        xyz, fps, out_xyz, nx);
    query_ball_kernel<<<NBATCH * NPOINT / 8, 256, 0, stream>>>(xyz, nx, g0, g1);

    Params36 P;
    for (int i = 0; i < 36; ++i) P.p[i] = (const float*)d_in[2 + i];
    fold_weights_kernel<<<1, 192, 0, stream>>>(P, Wf, bf);

    mlp_wmma_kernel<<<NBATCH * NPOINT / 8, 256, 0, stream>>>(
        g0, xyz, pts, nx, Wf, bf, out_pts, /*branch=*/0, /*K=*/16, /*mtiles=*/1);
    mlp_wmma_kernel<<<NBATCH * NPOINT / 8, 256, 0, stream>>>(
        g1, xyz, pts, nx, Wf, bf, out_pts, /*branch=*/1, /*K=*/32, /*mtiles=*/2);
}
